// TripletLoss_25451976196909
// MI455X (gfx1250) — matrix operand order repaired and checked by reading, not compiled, not run on hardware
//
#include <hip/hip_runtime.h>

// ---------------------------------------------------------------------------
// Triplet loss with batch-hard mining, B=8192 D=512 fp32.
// Gram matrix via bf16 split-precision WMMA (hi*hi + hi*lo + lo*hi in f32),
// mining fused into the GEMM epilogue, per-row results combined with
// monotone unsigned atomics on float bit patterns (distances are >= 0).
// ---------------------------------------------------------------------------

typedef __attribute__((ext_vector_type(16))) __bf16 v16bf;
typedef __attribute__((ext_vector_type(8)))  float  v8f;

#define MARGIN   0.3f
#define EPS_D2   1e-12f
#define INF_BITS 0x7F800000u

// ----- fp32 -> bf16 (round-to-nearest-even) helpers -------------------------
__device__ __forceinline__ unsigned short f32_to_bf16_rne(float x) {
    unsigned u = __float_as_uint(x);
    unsigned r = u + 0x7FFFu + ((u >> 16) & 1u);
    return (unsigned short)(r >> 16);
}
__device__ __forceinline__ float bf16_bits_to_f32(unsigned short h) {
    return __uint_as_float(((unsigned)h) << 16);
}

// ----- kernel 0: split e1/e2 into hi/lo bf16 planes -------------------------
__global__ __launch_bounds__(256) void split_bf16(
    const float* __restrict__ e1, const float* __restrict__ e2,
    unsigned short* __restrict__ a_hi, unsigned short* __restrict__ a_lo,
    unsigned short* __restrict__ b_hi, unsigned short* __restrict__ b_lo,
    int n) {
    int i = blockIdx.x * blockDim.x + threadIdx.x;
    if (i >= n) return;
    float x = e1[i];
    unsigned short xh = f32_to_bf16_rne(x);
    a_hi[i] = xh;
    a_lo[i] = f32_to_bf16_rne(x - bf16_bits_to_f32(xh));
    float y = e2[i];
    unsigned short yh = f32_to_bf16_rne(y);
    b_hi[i] = yh;
    b_lo[i] = f32_to_bf16_rne(y - bf16_bits_to_f32(yh));
}

// ----- kernel 1: row squared norms + init hard-pos/hard-neg accumulators ----
__global__ __launch_bounds__(256) void norms_init(
    const float* __restrict__ e1, const float* __restrict__ e2,
    float* __restrict__ sq1, float* __restrict__ sq2,
    unsigned* __restrict__ hp, unsigned* __restrict__ hn, int D) {
    int row  = blockIdx.x * 8 + (threadIdx.x >> 5);  // one wave32 per row
    int lane = threadIdx.x & 31;
    const float* r1 = e1 + (size_t)row * D;
    const float* r2 = e2 + (size_t)row * D;
    float s1 = 0.f, s2 = 0.f;
    for (int k = lane; k < D; k += 32) {
        float x = r1[k]; s1 += x * x;
        float y = r2[k]; s2 += y * y;
    }
#pragma unroll
    for (int off = 16; off > 0; off >>= 1) {
        s1 += __shfl_xor(s1, off, 32);
        s2 += __shfl_xor(s2, off, 32);
    }
    if (lane == 0) {
        sq1[row] = s1;
        sq2[row] = s2;
        hp[row]  = 0u;        // max-identity for non-negative distances
        hn[row]  = INF_BITS;  // min-identity
    }
}

// ----- WMMA fragment loads (layouts per CDNA5 ISA 7.12.2) -------------------
union Frag { v16bf v; uint4 q[2]; };

// A 16x32 bf16: lane = M (lane&15); lanes 0-15 own K {0..7,16..23},
// lanes 16-31 own K {8..15,24..31}  -> two contiguous 16B chunks per lane.
__device__ __forceinline__ v16bf load_frag_a(
    const unsigned short* __restrict__ p, int m, int kb, int ko, int D) {
    Frag f;
    const size_t base = (size_t)m * D + kb + ko;
    f.q[0] = *(const uint4*)(p + base);
    f.q[1] = *(const uint4*)(p + base + 16);
    return f.v;
}
// B 32x16 bf16 (computing A·B^T from row-major e2): lane = N (lane&15);
// lanes 0-15 own K {0..7,16..23}, lanes 16-31 own K {8..15,24..31} -> two 16B chunks.
__device__ __forceinline__ v16bf load_frag_b(
    const unsigned short* __restrict__ p, int n, int kb, int ko, int D) {
    Frag f;
    const size_t base = (size_t)n * D + kb + ko;
    f.q[0] = *(const uint4*)(p + base);
    f.q[1] = *(const uint4*)(p + base + 16);
    return f.v;
}

__device__ __forceinline__ v8f wmma_bf16(v16bf a, v16bf b, v8f c) {
    return __builtin_amdgcn_wmma_f32_16x16x32_bf16(
        false, a, false, b, (short)0, c, false, false);
}

// ----- kernel 2: fused GEMM + hard mining -----------------------------------
// Workgroup = 8 waves (2 M-waves x 4 N-waves); wave tile = 32x64 (2x4 WMMAs).
// Workgroup tile = 64 rows x 256 cols. Grid = (B/64, B/256).
__global__ __launch_bounds__(256) void hard_mine(
    const unsigned short* __restrict__ a_hi, const unsigned short* __restrict__ a_lo,
    const unsigned short* __restrict__ b_hi, const unsigned short* __restrict__ b_lo,
    const float* __restrict__ sq1, const float* __restrict__ sq2,
    const long long* __restrict__ labels,
    unsigned* __restrict__ hp, unsigned* __restrict__ hn, int D) {
    const int lane = threadIdx.x & 31;
    const int w    = threadIdx.x >> 5;
    const int wm   = w & 1;          // 0..1
    const int wn   = w >> 1;         // 0..3
    const int rowBase = blockIdx.x * 64  + wm * 32;
    const int colBase = blockIdx.y * 256 + wn * 64;
    const int half = lane >> 4;      // which 16-lane half
    const int nl   = lane & 15;

    const int aKo = half ? 8 : 0;    // A K-chunk offset per half
    const int bKo = half ? 8 : 0;    // B K-chunk offset per half
    const int am0 = rowBase + nl;    // A row for mt=0 (lane = M)
    const int am1 = rowBase + 16 + nl;
    int bn[4];
#pragma unroll
    for (int nt = 0; nt < 4; ++nt) bn[nt] = colBase + nt * 16 + nl;  // lane = N

    v8f c[2][4] = {};  // f32 accumulators

    for (int kb = 0; kb < D; kb += 32) {
        v16bf ah0 = load_frag_a(a_hi, am0, kb, aKo, D);
        v16bf ah1 = load_frag_a(a_hi, am1, kb, aKo, D);
        v16bf al0 = load_frag_a(a_lo, am0, kb, aKo, D);
        v16bf al1 = load_frag_a(a_lo, am1, kb, aKo, D);
        v16bf bh[4], bl[4];
#pragma unroll
        for (int nt = 0; nt < 4; ++nt) {
            bh[nt] = load_frag_b(b_hi, bn[nt], kb, bKo, D);
            bl[nt] = load_frag_b(b_lo, bn[nt], kb, bKo, D);
        }
#pragma unroll
        for (int nt = 0; nt < 4; ++nt) {
            // split-precision accumulate: hi*hi + hi*lo + lo*hi (drop lo*lo)
            c[0][nt] = wmma_bf16(ah0, bh[nt], c[0][nt]);
            c[0][nt] = wmma_bf16(ah0, bl[nt], c[0][nt]);
            c[0][nt] = wmma_bf16(al0, bh[nt], c[0][nt]);
            c[1][nt] = wmma_bf16(ah1, bh[nt], c[1][nt]);
            c[1][nt] = wmma_bf16(ah1, bl[nt], c[1][nt]);
            c[1][nt] = wmma_bf16(al1, bh[nt], c[1][nt]);
        }
    }

    // ---- epilogue: dist = sqrt(max(s1+s2-2G, eps)); masked row max/min ----
    int   l2v[4]; float s2v[4]; int nglob[4];
#pragma unroll
    for (int nt = 0; nt < 4; ++nt) {
        int n = colBase + nt * 16 + nl;      // C/D layout: lane&15 = N
        nglob[nt] = n;
        l2v[nt]   = (int)labels[n];
        s2v[nt]   = sq2[n];
    }

#pragma unroll
    for (int mt = 0; mt < 2; ++mt) {
#pragma unroll
        for (int i = 0; i < 8; ++i) {
            // C/D layout: VGPR i holds M=i (lanes 0-15) / M=i+8 (lanes 16-31)
            const int m  = rowBase + mt * 16 + half * 8 + i;
            const int l1 = (int)labels[m];
            const float s1 = sq1[m];
            unsigned posv = 0u, negv = INF_BITS;
#pragma unroll
            for (int nt = 0; nt < 4; ++nt) {
                float g  = c[mt][nt][i];
                float d2 = s1 + s2v[nt] - 2.0f * g;
                float d  = sqrtf(fmaxf(d2, EPS_D2));
                unsigned db = __float_as_uint(d);
                if (l1 == l2v[nt] && m != nglob[nt]) posv = posv > db ? posv : db;
                if (l1 != l2v[nt])                   negv = negv < db ? negv : db;
            }
            // reduce across the 16 lanes sharing this row (stay within half)
#pragma unroll
            for (int off = 1; off < 16; off <<= 1) {
                unsigned pp = __shfl_xor(posv, off, 32);
                unsigned nn = __shfl_xor(negv, off, 32);
                posv = posv > pp ? posv : pp;
                negv = negv < nn ? negv : nn;
            }
            if (nl == 0) {  // lanes 0 and 16 carry rows m and m+8
                atomicMax(&hp[m], posv);
                atomicMin(&hn[m], negv);
            }
        }
    }
}

// ----- kernel 3: final scalar reduction -------------------------------------
__global__ __launch_bounds__(256) void finalize(
    const unsigned* __restrict__ hp, const unsigned* __restrict__ hn,
    float* __restrict__ out, int B) {
    __shared__ float ssum[256];
    __shared__ float scnt[256];
    float sum = 0.f, cnt = 0.f;
    for (int r = threadIdx.x; r < B; r += 256) {
        unsigned p = hp[r], q = hn[r];
        if (p != 0u && q != INF_BITS) {       // row had a positive and a negative
            float l = __uint_as_float(p) - __uint_as_float(q) + MARGIN;
            sum += fmaxf(l, 0.f);
            cnt += 1.f;
        }
    }
    ssum[threadIdx.x] = sum;
    scnt[threadIdx.x] = cnt;
    __syncthreads();
    for (int s = 128; s > 0; s >>= 1) {
        if (threadIdx.x < s) {
            ssum[threadIdx.x] += ssum[threadIdx.x + s];
            scnt[threadIdx.x] += scnt[threadIdx.x + s];
        }
        __syncthreads();
    }
    if (threadIdx.x == 0)
        out[0] = (scnt[0] > 0.f) ? (ssum[0] / scnt[0]) : 0.f;
}

// ---------------------------------------------------------------------------
extern "C" void kernel_launch(void* const* d_in, const int* in_sizes, int n_in,
                              void* d_out, int out_size, void* d_ws, size_t ws_size,
                              hipStream_t stream) {
    const float*     e1     = (const float*)d_in[0];
    const float*     e2     = (const float*)d_in[1];
    const long long* labels = (const long long*)d_in[2];
    const int B = in_sizes[2];          // 8192
    const int D = in_sizes[0] / B;      // 512

    const size_t mat = (size_t)B * D * sizeof(unsigned short);  // 8 MB each
    char* ws = (char*)d_ws;
    unsigned short* a_hi = (unsigned short*)(ws);
    unsigned short* a_lo = (unsigned short*)(ws + mat);
    unsigned short* b_hi = (unsigned short*)(ws + 2 * mat);
    unsigned short* b_lo = (unsigned short*)(ws + 3 * mat);
    float*    sq1 = (float*)(ws + 4 * mat);
    float*    sq2 = sq1 + B;
    unsigned* hp  = (unsigned*)(sq2 + B);
    unsigned* hn  = hp + B;
    const size_t needed = 4 * mat + 4 * (size_t)B * sizeof(float);
    if (ws_size < needed) return;  // workspace too small to run

    const int n = B * D;
    split_bf16<<<(n + 255) / 256, 256, 0, stream>>>(e1, e2, a_hi, a_lo, b_hi, b_lo, n);
    norms_init<<<B / 8, 256, 0, stream>>>(e1, e2, sq1, sq2, hp, hn, D);
    dim3 grid(B / 64, B / 256);
    hard_mine<<<grid, 256, 0, stream>>>(a_hi, a_lo, b_hi, b_lo, sq1, sq2, labels, hp, hn, D);
    finalize<<<1, 256, 0, stream>>>(hp, hn, (float*)d_out, B);
}
